// Graphnetwork2_phonon_12111807775405
// MI455X (gfx1250) — matrix operand order repaired and checked by reading, not compiled, not run on hardware
//
#include <hip/hip_runtime.h>
#include <math.h>

// ---------------------------------------------------------------------------
// Types for CDNA5 WMMA (wave32, 16x16x32 f16 -> f32)
// ---------------------------------------------------------------------------
typedef _Float16 half_t;
typedef __attribute__((ext_vector_type(16))) _Float16 v16h;
typedef __attribute__((ext_vector_type(8)))  float    v8f;

__device__ __forceinline__ v8f wmma_f16(v16h a, v16h b, v8f c) {
    return __builtin_amdgcn_wmma_f32_16x16x32_f16(
        false, a, false, b, (short)0, c, false, false);
}

// A-fragment: 16x32 tile, row-major f16 source with leading dim ld.
// lane m = lane&15; halves i=0..15 hold K = (i/8)*16 + (lane>=16?8:0) + i%8
__device__ __forceinline__ v16h load_a_f16(const half_t* __restrict__ base,
                                           int ld, int lane, int k0) {
    const int m  = lane & 15;
    const int kb = (lane & 16) ? 8 : 0;
    const half_t* p = base + (long)m * ld + k0 + kb;
    union { v16h v; float4 q[2]; } u;
    u.q[0] = *(const float4*)(p);
    u.q[1] = *(const float4*)(p + 16);
    return u.v;
}

// B-fragment from pre-packed weights: frag = 32 lanes x 16 contiguous halves.
__device__ __forceinline__ v16h load_b_packed(const half_t* __restrict__ frag_base,
                                              int lane) {
    const half_t* p = frag_base + lane * 16;
    union { v16h v; float4 q[2]; } u;
    u.q[0] = *(const float4*)(p);
    u.q[1] = *(const float4*)(p + 8);
    return u.v;
}

// CDNA5 packed f16 atomic add (GLOBAL_ATOMIC_PK_ADD_F16, no return -> STOREcnt)
__device__ __forceinline__ void atomic_pk_add_f16(half_t* p, float v0, float v1) {
    union { half_t h[2]; unsigned int u; } pk;
    pk.h[0] = (half_t)v0;
    pk.h[1] = (half_t)v1;
    asm volatile("global_atomic_pk_add_f16 %0, %1, off"
                 : : "v"(p), "v"(pk.u) : "memory");
}

// ---------------------------------------------------------------------------
// Prep kernels
// ---------------------------------------------------------------------------
__global__ void convert_pad_f16(const float* __restrict__ src, half_t* __restrict__ dst,
                                int K, int Nd, int Kp, int Np) {
    long idx = (long)blockIdx.x * blockDim.x + threadIdx.x;
    long tot = (long)Kp * Np;
    if (idx >= tot) return;
    long k = idx / Np;
    int  n = (int)(idx - k * Np);
    float v = (k < K && n < Nd) ? src[k * (long)Nd + n] : 0.0f;
    dst[idx] = (half_t)v;
}

// Pack f32 W[K][Nd] (zero-padded to Kp x Np) into WMMA B-fragment order:
// frag (kt,nt): 512 halves; lane l holds K = kt*32 + (l>=16?8:0) + (j&7) + (j>=8?16:0), N = nt*16 + (l&15)
__global__ void pack_b_frags(const float* __restrict__ src, half_t* __restrict__ dst,
                             int K, int Nd, int Kp, int Np) {
    long idx = (long)blockIdx.x * blockDim.x + threadIdx.x;
    long tot = (long)Kp * Np;
    if (idx >= tot) return;
    const int nNT = Np >> 4;
    long frag  = idx >> 9;
    int within = (int)(idx & 511);
    int lane = within >> 4;
    int j    = within & 15;
    int kt = (int)(frag / nNT);
    int nt = (int)(frag - (long)kt * nNT);
    int k = kt * 32 + ((lane & 16) ? 8 : 0) + (j & 7) + ((j & 8) ? 16 : 0);
    int n = nt * 16 + (lane & 15);
    float v = (k < K && n < Nd) ? src[(long)k * Nd + n] : 0.0f;
    dst[idx] = (half_t)v;
}

__global__ void zero_f32(float* __restrict__ p, long count) {
    long i = (long)blockIdx.x * blockDim.x + threadIdx.x;
    if (i < count) p[i] = 0.0f;
}

// ---------------------------------------------------------------------------
// Encoder: h = prelu(xp @ W1 + b1) @ W2 + b2
// ---------------------------------------------------------------------------
__global__ __launch_bounds__(32) void enc_node_kernel(
    const half_t* __restrict__ xp,
    const half_t* __restrict__ W1p, const float* __restrict__ b1,
    const float* __restrict__ alpha_p,
    const half_t* __restrict__ W2p, const float* __restrict__ b2,
    half_t* __restrict__ h) {
    __shared__ __align__(16) half_t sMid[16 * 128];
    const int lane = threadIdx.x;
    const long m0  = (long)blockIdx.x * 16;
    const int nloc  = lane & 15;
    const int mbase = (lane & 16) ? 8 : 0;
    const float alpha = alpha_p[0];

    v16h afr[4];
#pragma unroll
    for (int kt = 0; kt < 4; ++kt) afr[kt] = load_a_f16(xp + m0 * 128, 128, lane, kt * 32);

    for (int nt = 0; nt < 8; ++nt) {
        v8f acc = {};
#pragma unroll
        for (int kt = 0; kt < 4; ++kt)
            acc = wmma_f16(afr[kt], load_b_packed(W1p + (((long)kt * 8 + nt) << 9), lane), acc);
        const int n = nt * 16 + nloc;
        const float bias = b1[n];
#pragma unroll
        for (int r = 0; r < 8; ++r) {
            float v = acc[r] + bias;
            v = v >= 0.0f ? v : alpha * v;
            sMid[(mbase + r) * 128 + n] = (half_t)v;
        }
    }
    __syncthreads();

    v16h a2[4];
#pragma unroll
    for (int kt = 0; kt < 4; ++kt) a2[kt] = load_a_f16(sMid, 128, lane, kt * 32);

    for (int nt = 0; nt < 8; ++nt) {
        v8f acc = {};
#pragma unroll
        for (int kt = 0; kt < 4; ++kt)
            acc = wmma_f16(a2[kt], load_b_packed(W2p + (((long)kt * 8 + nt) << 9), lane), acc);
        const int n = nt * 16 + nloc;
        const float bias = b2[n];
#pragma unroll
        for (int r = 0; r < 8; ++r)
            sMid[(mbase + r) * 128 + n] = (half_t)(acc[r] + bias);
    }
    __syncthreads();
    for (int idx = lane; idx < 16 * 16; idx += 32) {
        const int m = idx >> 4, s = idx & 15;
        *(float4*)(h + (m0 + m) * 128 + s * 8) = *(const float4*)(sMid + m * 128 + s * 8);
    }
}

// ---------------------------------------------------------------------------
// Edge encoder: e = prelu(e_raw@W1+b1)@W2+b2, e_raw computed inline (K=3 via VALU)
// ---------------------------------------------------------------------------
__global__ __launch_bounds__(32) void enc_edge_kernel(
    const float* __restrict__ edge_vec, const float* __restrict__ edge_shift,
    const float* __restrict__ W1 /*3x128 f32*/, const float* __restrict__ b1,
    const float* __restrict__ alpha_p,
    const half_t* __restrict__ W2p, const float* __restrict__ b2,
    half_t* __restrict__ e) {
    __shared__ float sEr[16][3];
    __shared__ __align__(16) half_t sMid[16 * 128];
    const int lane = threadIdx.x;
    const long e0  = (long)blockIdx.x * 16;
    const int nloc  = lane & 15;
    const int mbase = (lane & 16) ? 8 : 0;

    if (lane < 16) {
        const long eid = e0 + lane;
        const float vx = edge_vec[eid * 3 + 0];
        const float vy = edge_vec[eid * 3 + 1];
        const float vz = edge_vec[eid * 3 + 2];
        const float len = sqrtf(vx * vx + vy * vy + vz * vz);
        const float u = 2.0f * (len * 0.25f - 1.0f);
        float y = 0.5f * (1.0f - cosf(3.14159265358979323846f * u));
        if (u > 0.0f)  y = 0.0f;
        if (u < -1.0f) y = 1.0f;
        sEr[lane][0] = y * edge_shift[eid * 3 + 0];
        sEr[lane][1] = y * edge_shift[eid * 3 + 1];
        sEr[lane][2] = y * edge_shift[eid * 3 + 2];
    }
    __syncthreads();

    const float alpha = alpha_p[0];
    for (int idx = lane; idx < 16 * 128; idx += 32) {
        const int m = idx >> 7, n = idx & 127;
        float v = sEr[m][0] * W1[n] + sEr[m][1] * W1[128 + n] + sEr[m][2] * W1[256 + n] + b1[n];
        v = v >= 0.0f ? v : alpha * v;
        sMid[m * 128 + n] = (half_t)v;
    }
    __syncthreads();

    v16h a2[4];
#pragma unroll
    for (int kt = 0; kt < 4; ++kt) a2[kt] = load_a_f16(sMid, 128, lane, kt * 32);

    for (int nt = 0; nt < 8; ++nt) {
        v8f acc = {};
#pragma unroll
        for (int kt = 0; kt < 4; ++kt)
            acc = wmma_f16(a2[kt], load_b_packed(W2p + (((long)kt * 8 + nt) << 9), lane), acc);
        const int n = nt * 16 + nloc;
        const float bias = b2[n];
#pragma unroll
        for (int r = 0; r < 8; ++r)
            sMid[(mbase + r) * 128 + n] = (half_t)(acc[r] + bias);
    }
    __syncthreads();
    for (int idx = lane; idx < 16 * 16; idx += 32) {
        const int m = idx >> 4, s = idx & 15;
        *(float4*)(e + (e0 + m) * 128 + s * 8) = *(const float4*)(sMid + m * 128 + s * 8);
    }
}

// ---------------------------------------------------------------------------
// Edge model (dominant), M=32 edges per wave: each B fragment feeds 2 WMMAs.
// gather -> GEMM(384->256) -> LN -> prelu -> GEMM(256->128)
// residual e += e_new (float4 stores), scatter e_new into agg[col] (pk_add_f16)
// ---------------------------------------------------------------------------
__global__ __launch_bounds__(32) void edge_layer_kernel(
    const half_t* __restrict__ h, half_t* __restrict__ e,
    const int* __restrict__ erow, const int* __restrict__ ecol,
    const half_t* __restrict__ W1p, const float* __restrict__ b1,
    const float* __restrict__ lng, const float* __restrict__ lnb,
    const float* __restrict__ alpha_p,
    const half_t* __restrict__ W2p, const float* __restrict__ b2,
    half_t* __restrict__ agg) {
    __shared__ __align__(16) half_t sA[2][16 * 256];  // [h[row]|h[col]], reused as midh
    __shared__ __align__(16) half_t sE[2][16 * 128];  // e tiles
    __shared__ __align__(16) float  sMid[2][16 * 256];
    __shared__ float sMean[32], sRstd[32];
    __shared__ int sSrc[32], sDst[32];
    const int lane = threadIdx.x;
    const long e0  = (long)blockIdx.x * 32;
    const int nloc  = lane & 15;
    const int mbase = (lane & 16) ? 8 : 0;
    const float alpha = alpha_p[0];

    sSrc[lane] = erow[e0 + lane];
    sDst[lane] = ecol[e0 + lane];
    __syncthreads();

    for (int idx = lane; idx < 32 * 16; idx += 32) {
        const int m = idx >> 4, s = idx & 15;   // m: 0..31, s: 16B segment
        const int t = m >> 4, ml = m & 15;
        *(float4*)(&sA[t][ml * 256 + s * 8])       = *(const float4*)(h + (long)sSrc[m] * 128 + s * 8);
        *(float4*)(&sA[t][ml * 256 + 128 + s * 8]) = *(const float4*)(h + (long)sDst[m] * 128 + s * 8);
        *(float4*)(&sE[t][ml * 128 + s * 8])       = *(const float4*)(e + (e0 + m) * 128 + s * 8);
    }
    __syncthreads();

    v16h a0[12], a1[12];
#pragma unroll
    for (int kt = 0; kt < 8; ++kt) {
        a0[kt] = load_a_f16(sA[0], 256, lane, kt * 32);
        a1[kt] = load_a_f16(sA[1], 256, lane, kt * 32);
    }
#pragma unroll
    for (int kt = 0; kt < 4; ++kt) {
        a0[8 + kt] = load_a_f16(sE[0], 128, lane, kt * 32);
        a1[8 + kt] = load_a_f16(sE[1], 128, lane, kt * 32);
    }

    // GEMM1: 32x384 @ 384x256, depth-1 B prefetch, B reused by both tiles
    for (int nt = 0; nt < 16; ++nt) {
        v8f acc0 = {}, acc1 = {};
        v16h b = load_b_packed(W1p + ((long)nt << 9), lane);
#pragma unroll
        for (int kt = 0; kt < 12; ++kt) {
            v16h bn = b;
            if (kt < 11) bn = load_b_packed(W1p + (((long)(kt + 1) * 16 + nt) << 9), lane);
            acc0 = wmma_f16(a0[kt], b, acc0);
            acc1 = wmma_f16(a1[kt], b, acc1);
            b = bn;
        }
        const int n = nt * 16 + nloc;
        const float bias = b1[n];
#pragma unroll
        for (int r = 0; r < 8; ++r) {
            sMid[0][(mbase + r) * 256 + n] = acc0[r] + bias;
            sMid[1][(mbase + r) * 256 + n] = acc1[r] + bias;
        }
    }
    __syncthreads();

    // LayerNorm stats: 32 rows, one per lane
    {
        const int t = lane >> 4, ml = lane & 15;
        const float4* rp = (const float4*)(&sMid[t][ml * 256]);
        float s = 0.0f, s2 = 0.0f;
        for (int c = 0; c < 64; ++c) {
            float4 q = rp[c];
            s  += q.x + q.y + q.z + q.w;
            s2 += q.x * q.x + q.y * q.y + q.z * q.z + q.w * q.w;
        }
        const float mean = s * (1.0f / 256.0f);
        const float var  = s2 * (1.0f / 256.0f) - mean * mean;
        sMean[lane] = mean;
        sRstd[lane] = rsqrtf(var + 1e-5f);
    }
    __syncthreads();

    for (int idx = lane; idx < 32 * 256; idx += 32) {
        const int m = idx >> 8, c = idx & 255;
        const int t = m >> 4, ml = m & 15;
        float v = (sMid[t][ml * 256 + c] - sMean[m]) * sRstd[m] * lng[c] + lnb[c];
        v = v >= 0.0f ? v : alpha * v;
        sA[t][ml * 256 + c] = (half_t)v;   // reuse sA as midh (a0/a1 in regs)
    }
    __syncthreads();

    v16h c0[8], c1[8];
#pragma unroll
    for (int kt = 0; kt < 8; ++kt) {
        c0[kt] = load_a_f16(sA[0], 256, lane, kt * 32);
        c1[kt] = load_a_f16(sA[1], 256, lane, kt * 32);
    }

    // GEMM2: 32x256 @ 256x128 -> sMid[t][0:16*128]
    for (int nt = 0; nt < 8; ++nt) {
        v8f acc0 = {}, acc1 = {};
        v16h b = load_b_packed(W2p + ((long)nt << 9), lane);
#pragma unroll
        for (int kt = 0; kt < 8; ++kt) {
            v16h bn = b;
            if (kt < 7) bn = load_b_packed(W2p + (((long)(kt + 1) * 8 + nt) << 9), lane);
            acc0 = wmma_f16(c0[kt], b, acc0);
            acc1 = wmma_f16(c1[kt], b, acc1);
            b = bn;
        }
        const int n = nt * 16 + nloc;
        const float bias = b2[n];
#pragma unroll
        for (int r = 0; r < 8; ++r) {
            sMid[0][(mbase + r) * 128 + n] = acc0[r] + bias;
            sMid[1][(mbase + r) * 128 + n] = acc1[r] + bias;
        }
    }
    __syncthreads();

    // Epilogue: e += e_new (float4 stores), agg[col] += e_new (packed f16 atomics)
    for (int idx = lane; idx < 32 * 16; idx += 32) {
        const int m = idx >> 4, s = idx & 15;
        const int t = m >> 4, ml = m & 15;
        const float*  mp = &sMid[t][ml * 128 + s * 8];
        const half_t* eo = &sE[t][ml * 128 + s * 8];
        union { half_t hh[8]; float4 q; } o;
        float vv[8];
#pragma unroll
        for (int j = 0; j < 8; ++j) {
            vv[j]   = mp[j];
            o.hh[j] = (half_t)((float)eo[j] + vv[j]);
        }
        *(float4*)(e + (e0 + m) * 128 + s * 8) = o.q;
        half_t* ap = agg + (long)sDst[m] * 128 + s * 8;
#pragma unroll
        for (int j = 0; j < 4; ++j)
            atomic_pk_add_f16(ap + j * 2, vv[2 * j], vv[2 * j + 1]);
    }
}

// ---------------------------------------------------------------------------
// Node model: [h | agg] -> GEMM(256->256) -> LN -> prelu -> GEMM(256->128); h += new
// ---------------------------------------------------------------------------
__global__ __launch_bounds__(32) void node_layer_kernel(
    half_t* __restrict__ h, const half_t* __restrict__ agg,
    const half_t* __restrict__ W1p, const float* __restrict__ b1,
    const float* __restrict__ lng, const float* __restrict__ lnb,
    const float* __restrict__ alpha_p,
    const half_t* __restrict__ W2p, const float* __restrict__ b2) {
    __shared__ __align__(16) half_t sA[16 * 256];
    __shared__ __align__(16) float sMid[16 * 256];
    __shared__ float sMean[16], sRstd[16];
    const int lane = threadIdx.x;
    const long m0  = (long)blockIdx.x * 16;
    const int nloc  = lane & 15;
    const int mbase = (lane & 16) ? 8 : 0;

    for (int idx = lane; idx < 16 * 16; idx += 32) {
        const int m = idx >> 4, s = idx & 15;
        *(float4*)(sA + m * 256 + s * 8)       = *(const float4*)(h + (m0 + m) * 128 + s * 8);
        *(float4*)(sA + m * 256 + 128 + s * 8) = *(const float4*)(agg + (m0 + m) * 128 + s * 8);
    }
    __syncthreads();

    v16h afr[8];
#pragma unroll
    for (int kt = 0; kt < 8; ++kt) afr[kt] = load_a_f16(sA, 256, lane, kt * 32);

    for (int nt = 0; nt < 16; ++nt) {
        v8f acc = {};
        v16h b = load_b_packed(W1p + ((long)nt << 9), lane);
#pragma unroll
        for (int kt = 0; kt < 8; ++kt) {
            v16h bn = b;
            if (kt < 7) bn = load_b_packed(W1p + (((long)(kt + 1) * 16 + nt) << 9), lane);
            acc = wmma_f16(afr[kt], b, acc);
            b = bn;
        }
        const int n = nt * 16 + nloc;
        const float bias = b1[n];
#pragma unroll
        for (int r = 0; r < 8; ++r)
            sMid[(mbase + r) * 256 + n] = acc[r] + bias;
    }
    __syncthreads();

    if (lane < 16) {
        const float4* rp = (const float4*)(sMid + lane * 256);
        float s = 0.0f, s2 = 0.0f;
        for (int c = 0; c < 64; ++c) {
            float4 q = rp[c];
            s  += q.x + q.y + q.z + q.w;
            s2 += q.x * q.x + q.y * q.y + q.z * q.z + q.w * q.w;
        }
        const float mean = s * (1.0f / 256.0f);
        const float var  = s2 * (1.0f / 256.0f) - mean * mean;
        sMean[lane] = mean;
        sRstd[lane] = rsqrtf(var + 1e-5f);
    }
    __syncthreads();

    const float alpha = alpha_p[0];
    for (int idx = lane; idx < 16 * 256; idx += 32) {
        const int m = idx >> 8, c = idx & 255;
        float v = (sMid[idx] - sMean[m]) * sRstd[m] * lng[c] + lnb[c];
        v = v >= 0.0f ? v : alpha * v;
        sA[m * 256 + c] = (half_t)v;
    }
    __syncthreads();

    v16h a2[8];
#pragma unroll
    for (int kt = 0; kt < 8; ++kt) a2[kt] = load_a_f16(sA, 256, lane, kt * 32);

    for (int nt = 0; nt < 8; ++nt) {
        v8f acc = {};
        v16h b = load_b_packed(W2p + ((long)nt << 9), lane);
#pragma unroll
        for (int kt = 0; kt < 8; ++kt) {
            v16h bn = b;
            if (kt < 7) bn = load_b_packed(W2p + (((long)(kt + 1) * 8 + nt) << 9), lane);
            acc = wmma_f16(a2[kt], b, acc);
            b = bn;
        }
        const int n = nt * 16 + nloc;
        const float bias = b2[n];
#pragma unroll
        for (int r = 0; r < 8; ++r)
            sMid[(mbase + r) * 128 + n] = acc[r] + bias;
    }
    __syncthreads();

    for (int idx = lane; idx < 16 * 16; idx += 32) {
        const int m = idx >> 4, s = idx & 15;
        const float* mp = sMid + m * 128 + s * 8;
        half_t* hp = h + (m0 + m) * 128 + s * 8;
        float4 oldq = *(const float4*)hp;
        union { half_t hh[8]; float4 q; } o;
        const half_t* oldh = (const half_t*)&oldq;
#pragma unroll
        for (int j = 0; j < 8; ++j)
            o.hh[j] = (half_t)((float)oldh[j] + mp[j]);
        *(float4*)hp = o.q;
    }
}

// ---------------------------------------------------------------------------
// Output head: d1 = prelu(LN(h@W1+b1))@W2+b2 ; out[batch[n]] += d1
// ---------------------------------------------------------------------------
__global__ __launch_bounds__(32) void out_head_kernel(
    const half_t* __restrict__ h,
    const half_t* __restrict__ W1p, const float* __restrict__ b1,
    const float* __restrict__ lng, const float* __restrict__ lnb,
    const float* __restrict__ alpha_p,
    const half_t* __restrict__ W2p, const float* __restrict__ b2,
    const int* __restrict__ batch, float* __restrict__ out) {
    __shared__ __align__(16) float sMid[16 * 128];
    __shared__ __align__(16) half_t sMidH[16 * 128];
    __shared__ float sMean[16], sRstd[16];
    __shared__ int sB[16];
    const int lane = threadIdx.x;
    const long m0  = (long)blockIdx.x * 16;
    const int nloc  = lane & 15;
    const int mbase = (lane & 16) ? 8 : 0;

    if (lane < 16) sB[lane] = batch[m0 + lane];

    v16h afr[4];
#pragma unroll
    for (int kt = 0; kt < 4; ++kt) afr[kt] = load_a_f16(h + m0 * 128, 128, lane, kt * 32);

    for (int nt = 0; nt < 8; ++nt) {
        v8f acc = {};
#pragma unroll
        for (int kt = 0; kt < 4; ++kt)
            acc = wmma_f16(afr[kt], load_b_packed(W1p + (((long)kt * 8 + nt) << 9), lane), acc);
        const int n = nt * 16 + nloc;
        const float bias = b1[n];
#pragma unroll
        for (int r = 0; r < 8; ++r)
            sMid[(mbase + r) * 128 + n] = acc[r] + bias;
    }
    __syncthreads();

    if (lane < 16) {
        const float4* rp = (const float4*)(sMid + lane * 128);
        float s = 0.0f, s2 = 0.0f;
        for (int c = 0; c < 32; ++c) {
            float4 q = rp[c];
            s  += q.x + q.y + q.z + q.w;
            s2 += q.x * q.x + q.y * q.y + q.z * q.z + q.w * q.w;
        }
        const float mean = s * (1.0f / 128.0f);
        const float var  = s2 * (1.0f / 128.0f) - mean * mean;
        sMean[lane] = mean;
        sRstd[lane] = rsqrtf(var + 1e-5f);
    }
    __syncthreads();

    const float alpha = alpha_p[0];
    for (int idx = lane; idx < 16 * 128; idx += 32) {
        const int m = idx >> 7, c = idx & 127;
        float v = (sMid[idx] - sMean[m]) * sRstd[m] * lng[c] + lnb[c];
        v = v >= 0.0f ? v : alpha * v;
        sMidH[m * 128 + c] = (half_t)v;
    }
    __syncthreads();

    v16h a2[4];
#pragma unroll
    for (int kt = 0; kt < 4; ++kt) a2[kt] = load_a_f16(sMidH, 128, lane, kt * 32);

    for (int nt = 0; nt < 4; ++nt) {
        v8f acc = {};
#pragma unroll
        for (int kt = 0; kt < 4; ++kt)
            acc = wmma_f16(a2[kt], load_b_packed(W2p + (((long)kt * 4 + nt) << 9), lane), acc);
        const int n = nt * 16 + nloc;
        if (n < 51) {
            const float bias = b2[n];
#pragma unroll
            for (int r = 0; r < 8; ++r)
                atomicAdd(out + (long)sB[mbase + r] * 51 + n, acc[r] + bias);
        }
    }
}

// ---------------------------------------------------------------------------
// Host-side orchestration
// ---------------------------------------------------------------------------
extern "C" void kernel_launch(void* const* d_in, const int* in_sizes, int n_in,
                              void* d_out, int out_size, void* d_ws, size_t ws_size,
                              hipStream_t stream) {
    (void)in_sizes; (void)n_in; (void)out_size; (void)ws_size;
    constexpr int N = 40000, E = 640000, G = 256, L = 3, DOUT = 51;

    const float* x          = (const float*)d_in[0];
    const float* edge_vec   = (const float*)d_in[1];
    const float* edge_shift = (const float*)d_in[2];
    const int*   edge_index = (const int*)d_in[3];
    const int*   batch      = (const int*)d_in[4];
    const float* enc_node_W1 = (const float*)d_in[6];
    const float* enc_node_b1 = (const float*)d_in[7];
    const float* enc_node_a  = (const float*)d_in[8];
    const float* enc_node_W2 = (const float*)d_in[9];
    const float* enc_node_b2 = (const float*)d_in[10];
    const float* enc_edge_W1 = (const float*)d_in[11];
    const float* enc_edge_b1 = (const float*)d_in[12];
    const float* enc_edge_a  = (const float*)d_in[13];
    const float* enc_edge_W2 = (const float*)d_in[14];
    const float* enc_edge_b2 = (const float*)d_in[15];
    const float* edge_W1   = (const float*)d_in[16];
    const float* edge_b1   = (const float*)d_in[17];
    const float* edge_ln_g = (const float*)d_in[18];
    const float* edge_ln_b = (const float*)d_in[19];
    const float* edge_a    = (const float*)d_in[20];
    const float* edge_W2   = (const float*)d_in[21];
    const float* edge_b2   = (const float*)d_in[22];
    const float* node_W1   = (const float*)d_in[23];
    const float* node_b1   = (const float*)d_in[24];
    const float* node_ln_g = (const float*)d_in[25];
    const float* node_ln_b = (const float*)d_in[26];
    const float* node_a    = (const float*)d_in[27];
    const float* node_W2   = (const float*)d_in[28];
    const float* node_b2   = (const float*)d_in[29];
    const float* out_W1   = (const float*)d_in[30];
    const float* out_b1   = (const float*)d_in[31];
    const float* out_ln_g = (const float*)d_in[32];
    const float* out_ln_b = (const float*)d_in[33];
    const float* out_a    = (const float*)d_in[34];
    const float* out_W2   = (const float*)d_in[35];
    const float* out_b2   = (const float*)d_in[36];
    float* out = (float*)d_out;

    char* ws = (char*)d_ws;
    size_t off = 0;
    auto take = [&](size_t bytes) -> char* {
        char* p = ws + off;
        off += (bytes + 255) & ~(size_t)255;
        return p;
    };
    half_t* hbuf = (half_t*)take((size_t)N * 128 * 2);
    half_t* ebuf = (half_t*)take((size_t)E * 128 * 2);
    half_t* agg  = (half_t*)take((size_t)N * 128 * 2);
    half_t* xp   = (half_t*)take((size_t)N * 128 * 2);
    half_t* wEncN1 = (half_t*)take(128 * 128 * 2);
    half_t* wEncN2 = (half_t*)take(128 * 128 * 2);
    half_t* wEncE2 = (half_t*)take(128 * 128 * 2);
    half_t* wEdge1 = (half_t*)take((size_t)L * 384 * 256 * 2);
    half_t* wEdge2 = (half_t*)take((size_t)L * 256 * 128 * 2);
    half_t* wNode1 = (half_t*)take((size_t)L * 256 * 256 * 2);
    half_t* wNode2 = (half_t*)take((size_t)L * 256 * 128 * 2);
    half_t* wOut1  = (half_t*)take(128 * 128 * 2);
    half_t* wOut2  = (half_t*)take(128 * 64 * 2);

    auto pack = [&](const float* s, half_t* d, int K, int Nd, int Kp, int Np) {
        long tot = (long)Kp * Np;
        pack_b_frags<<<dim3((unsigned)((tot + 255) / 256)), dim3(256), 0, stream>>>(
            s, d, K, Nd, Kp, Np);
    };
    pack(enc_node_W1, wEncN1, 118, 128, 128, 128);
    pack(enc_node_W2, wEncN2, 128, 128, 128, 128);
    pack(enc_edge_W2, wEncE2, 128, 128, 128, 128);
    pack(edge_W1, wEdge1, L * 384, 256, L * 384, 256);
    pack(edge_W2, wEdge2, L * 256, 128, L * 256, 128);
    pack(node_W1, wNode1, L * 256, 256, L * 256, 256);
    pack(node_W2, wNode2, L * 256, 128, L * 256, 128);
    pack(out_W1, wOut1, 128, 128, 128, 128);
    pack(out_W2, wOut2, 128, 51, 128, 64);
    {
        long tot = (long)N * 128;
        convert_pad_f16<<<dim3((unsigned)((tot + 255) / 256)), dim3(256), 0, stream>>>(
            x, xp, N, 118, N, 128);
    }

    const int* erow = edge_index;
    const int* ecol = edge_index + E;

    enc_node_kernel<<<dim3(N / 16), dim3(32), 0, stream>>>(
        xp, wEncN1, enc_node_b1, enc_node_a, wEncN2, enc_node_b2, hbuf);
    enc_edge_kernel<<<dim3(E / 16), dim3(32), 0, stream>>>(
        edge_vec, edge_shift, enc_edge_W1, enc_edge_b1, enc_edge_a,
        wEncE2, enc_edge_b2, ebuf);

    for (int i = 0; i < L; ++i) {
        long aggF32 = (long)N * 128 / 2;
        zero_f32<<<dim3((unsigned)((aggF32 + 255) / 256)), dim3(256), 0, stream>>>(
            (float*)agg, aggF32);
        edge_layer_kernel<<<dim3(E / 32), dim3(32), 0, stream>>>(
            hbuf, ebuf, erow, ecol,
            wEdge1 + (size_t)i * 384 * 256, edge_b1 + i * 256,
            edge_ln_g + i * 256, edge_ln_b + i * 256, edge_a + i,
            wEdge2 + (size_t)i * 256 * 128, edge_b2 + i * 128, agg);
        node_layer_kernel<<<dim3(N / 16), dim3(32), 0, stream>>>(
            hbuf, agg,
            wNode1 + (size_t)i * 256 * 256, node_b1 + i * 256,
            node_ln_g + i * 256, node_ln_b + i * 256, node_a + i,
            wNode2 + (size_t)i * 256 * 128, node_b2 + i * 128);
    }

    long outElems = (long)G * DOUT;
    zero_f32<<<dim3((unsigned)((outElems + 255) / 256)), dim3(256), 0, stream>>>(out, outElems);
    out_head_kernel<<<dim3(N / 16), dim3(32), 0, stream>>>(
        hbuf, wOut1, out_b1, out_ln_g, out_ln_b, out_a, wOut2, out_b2, batch, out);
}
